// MoleculeRegressor_4681514353046
// MI455X (gfx1250) — matrix operand order repaired and checked by reading, not compiled, not run on hardware
//
#include <hip/hip_runtime.h>
#include <hip/hip_bf16.h>
#include <cstdint>
#include <cstddef>

#define NNODES  50000
#define NEDGES  1600000
#define NGRAPHS 1024
#define HID     64
#define EDGE_F  16
#define IN_PAD  160   // 145 padded to 5 K-tiles of 32
#define NCONVS  3

typedef __attribute__((ext_vector_type(16))) __bf16 v16bf;
typedef __attribute__((ext_vector_type(8)))  float  v8f;

struct Pack32 { uint4 lo; uint4 hi; };

__device__ __forceinline__ v16bf make_frag(const uint4& lo, const uint4& hi) {
    Pack32 p{lo, hi};
    return __builtin_bit_cast(v16bf, p);
}

__device__ __forceinline__ unsigned short f2bf(float f) {
    unsigned u = __float_as_uint(f);
    unsigned r = u + 0x7FFFu + ((u >> 16) & 1u);   // round-to-nearest-even
    return (unsigned short)(r >> 16);
}

__device__ __forceinline__ float silu(float x) {
    return x / (1.0f + __expf(-x));
}

// ---------------------------------------------------------------- utilities
__global__ void zero_f32_kernel(float* p, int n) {
    int i = blockIdx.x * blockDim.x + threadIdx.x;
    if (i < n) p[i] = 0.0f;
}

__global__ void f32_to_bf16_kernel(const float* __restrict__ in,
                                   unsigned short* __restrict__ out, int n) {
    int i = blockIdx.x * blockDim.x + threadIdx.x;
    if (i < n) out[i] = f2bf(in[i]);
}

// Wm [NCONVS][145][HID] fp32  ->  Wt [NCONVS][HID][IN_PAD] bf16 (transposed, zero-padded)
__global__ void transform_weights_kernel(const float* __restrict__ Wm,
                                         unsigned short* __restrict__ Wt) {
    int idx = blockIdx.x * blockDim.x + threadIdx.x;       // (c*64 + n)*160 + k
    if (idx >= NCONVS * HID * IN_PAD) return;
    int k   = idx % IN_PAD;
    int n   = (idx / IN_PAD) % HID;
    int c   = idx / (IN_PAD * HID);
    Wt[idx] = (k < 145) ? f2bf(Wm[((size_t)c * 145 + k) * HID + n]) : (unsigned short)0;
}

// ---------------------------------------------------- edge message + scatter
// One wave = one 16-edge tile. 8 waves / block. grid = NEDGES/16/8 = 12500 exact.
__global__ void __launch_bounds__(256)
edge_msg_kernel(const unsigned short* __restrict__ hbf,   // [NNODES][64] bf16
                const float* __restrict__ pos,            // [NNODES][3]
                const float* __restrict__ eattr,          // [NEDGES][16]
                const int*   __restrict__ src,
                const int*   __restrict__ dst,
                const unsigned short* __restrict__ Wt,    // [64][160] bf16 (transposed)
                const float* __restrict__ bm,             // [64]
                float* __restrict__ agg)                  // [NNODES][64]
{
    __shared__ alignas(16) unsigned short At[8][16][IN_PAD];   // 40 KB

    const int wave = threadIdx.x >> 5;
    const int lane = threadIdx.x & 31;
    const int tile = blockIdx.x * 8 + wave;                    // < 100000 exactly
    const int e0   = tile * 16;

    // ---- stage h[src] | h[dst] rows (cols 0..127), coalesced ushort2 loads
    for (int r = 0; r < 16; ++r) {
        int e = e0 + r;
        int s = src[e], d = dst[e];
        ((ushort2*)&At[wave][r][0])[lane]  = ((const ushort2*)(hbf + (size_t)s * HID))[lane];
        ((ushort2*)&At[wave][r][HID])[lane] = ((const ushort2*)(hbf + (size_t)d * HID))[lane];
    }
    // ---- edge_attr (cols 129..144), 256 elems over 32 lanes
    for (int t = 0; t < 8; ++t) {
        int flat = t * 32 + lane;
        int rr = flat >> 4, cc = flat & 15;
        At[wave][rr][129 + cc] = f2bf(eattr[(size_t)(e0 + rr) * EDGE_F + cc]);
    }
    // ---- dij (col 128) + zero pad (cols 145..159)
    if (lane < 16) {
        int e = e0 + lane;
        int s = src[e], d = dst[e];
        float dx = pos[s * 3 + 0] - pos[d * 3 + 0];
        float dy = pos[s * 3 + 1] - pos[d * 3 + 1];
        float dz = pos[s * 3 + 2] - pos[d * 3 + 2];
        float dij = sqrtf(fmaxf(dx * dx + dy * dy + dz * dz, 1e-8f));
        At[wave][lane][128] = f2bf(dij);
        for (int c = 145; c < IN_PAD; ++c) At[wave][lane][c] = 0;
    }
    __syncthreads();

    // ---- A fragments: 16-bit A 16x32, contiguous k order matching B
    // lane<16: M=lane,    K = kb..kb+15  (kb = 32*kt)
    // lane>=16: M=lane-16, K = kb+16..kb+31
    const uint* rowp = (const uint*)&At[wave][lane & 15][0];
    const int khalf  = (lane >> 4) << 4;                      // 0 or 16
    v16bf afrag[5];
    #pragma unroll
    for (int kt = 0; kt < 5; ++kt) {
        int kb = kt * 32 + khalf;
        uint4 lo = *(const uint4*)(rowp + (kb >> 1));
        uint4 hi = *(const uint4*)(rowp + ((kb + 8) >> 1));
        afrag[kt] = make_frag(lo, hi);
    }

    // dst rows for this lane's C-layout half (m = v + 8*(lane>=16))
    const int mbase = (lane < 16) ? 0 : 8;
    int drow[8];
    #pragma unroll
    for (int v = 0; v < 8; ++v) drow[v] = dst[e0 + mbase + v];

    const int ncol = lane & 15;
    #pragma unroll
    for (int nt = 0; nt < 4; ++nt) {
        v8f acc = {};
        #pragma unroll
        for (int kt = 0; kt < 5; ++kt) {
            // B 32x16 bf16 layout: lane holds column n=lane&15, K = 16 contiguous
            int kb = kt * 32 + ((lane < 16) ? 0 : 16);
            const uint4* bp = (const uint4*)(Wt + (size_t)(nt * 16 + ncol) * IN_PAD + kb);
            v16bf bfrag = make_frag(bp[0], bp[1]);
            acc = __builtin_amdgcn_wmma_f32_16x16x32_bf16(
                      false, afrag[kt], false, bfrag, (short)0, acc, false, false);
        }
        float bias = bm[nt * 16 + ncol];
        #pragma unroll
        for (int v = 0; v < 8; ++v) {
            float m = silu(acc[v] + bias);
            atomicAdd(&agg[(size_t)drow[v] * HID + nt * 16 + ncol], m);
        }
    }
}

// ------------------------------------------------- node update + layernorm
// One thread per node; weights are lane-uniform -> scalar loads.
__global__ void node_update_kernel(float* __restrict__ agg,      // in: sum m; reused as u
                                   const float* __restrict__ Wu, // [64][64]
                                   const float* __restrict__ bu,
                                   const float* __restrict__ g,
                                   const float* __restrict__ b,
                                   float* __restrict__ hout, int n_nodes)
{
    int node = blockIdx.x * blockDim.x + threadIdx.x;
    if (node >= n_nodes) return;
    float* row = agg + (size_t)node * HID;
    float r[HID];
    #pragma unroll 8
    for (int k = 0; k < HID; ++k) r[k] = row[k] * 0.1f;
    float s1 = 0.0f, s2 = 0.0f;
    for (int j = 0; j < HID; ++j) {
        float a = bu[j];
        #pragma unroll 8
        for (int k = 0; k < HID; ++k) a += r[k] * Wu[k * HID + j];
        float x = silu(a);
        s1 += x; s2 += x * x;
        row[j] = x;                       // stash u in place
    }
    float mu  = s1 * (1.0f / HID);
    float var = s2 * (1.0f / HID) - mu * mu;
    float inv = rsqrtf(var + 1e-5f);
    for (int j = 0; j < HID; ++j)
        hout[(size_t)node * HID + j] = g[j] * (row[j] - mu) * inv + b[j];
}

// --------------------------------------------------------------- readout
__global__ void graph_scatter_kernel(const float* __restrict__ h,
                                     const int* __restrict__ gid,
                                     float* __restrict__ gf) {
    int idx = blockIdx.x * blockDim.x + threadIdx.x;    // over NNODES*64
    if (idx >= NNODES * HID) return;
    int node = idx >> 6, c = idx & 63;
    atomicAdd(&gf[(size_t)gid[node] * HID + c], h[idx]);
}

__global__ void head_kernel(const float* __restrict__ gf,
                            const float* __restrict__ Wh1,  // [64][32]
                            const float* __restrict__ bh1,
                            const float* __restrict__ Wh2,  // [32]
                            const float* __restrict__ bh2,
                            float* __restrict__ out) {
    int gidx = blockIdx.x * blockDim.x + threadIdx.x;
    if (gidx >= NGRAPHS) return;
    float r[HID];
    #pragma unroll 8
    for (int k = 0; k < HID; ++k) r[k] = gf[(size_t)gidx * HID + k] * 0.1f;
    float s = 0.0f;
    for (int j = 0; j < 32; ++j) {
        float a = bh1[j];
        #pragma unroll 8
        for (int k = 0; k < HID; ++k) a += r[k] * Wh1[k * 32 + j];
        s += silu(a) * Wh2[j];
    }
    out[gidx] = s + bh2[0];
}

// ---------------------------------------------------------------- launcher
extern "C" void kernel_launch(void* const* d_in, const int* in_sizes, int n_in,
                              void* d_out, int out_size, void* d_ws, size_t ws_size,
                              hipStream_t stream) {
    const float* node_attr = (const float*)d_in[0];
    const float* pos       = (const float*)d_in[1];
    const float* edge_attr = (const float*)d_in[2];
    const int*   src       = (const int*)  d_in[3];
    const int*   dst       = (const int*)  d_in[4];
    const int*   graph_ids = (const int*)  d_in[5];
    const float* Wm        = (const float*)d_in[6];
    const float* bm        = (const float*)d_in[7];
    const float* Wu        = (const float*)d_in[8];
    const float* bu        = (const float*)d_in[9];
    const float* ln_g      = (const float*)d_in[10];
    const float* ln_b      = (const float*)d_in[11];
    const float* Wh1       = (const float*)d_in[12];
    const float* bh1       = (const float*)d_in[13];
    const float* Wh2       = (const float*)d_in[14];
    const float* bh2       = (const float*)d_in[15];
    float* out = (float*)d_out;

    char* ws = (char*)d_ws;
    float*          h_f32 = (float*)(ws + 0);                    // 12,800,000 B
    unsigned short* h_bf  = (unsigned short*)(ws + 12800000);    //  6,400,000 B
    float*          agg   = (float*)(ws + 19200000);             // 12,800,000 B
    float*          gf    = (float*)(ws + 32000000);             //    262,144 B
    unsigned short* Wt    = (unsigned short*)(ws + 32262400);    //     61,440 B

    const int NH = NNODES * HID;            // 3,200,000

    transform_weights_kernel<<<(NCONVS * HID * IN_PAD + 255) / 256, 256, 0, stream>>>(Wm, Wt);

    for (int i = 0; i < NCONVS; ++i) {
        zero_f32_kernel<<<(NH + 255) / 256, 256, 0, stream>>>(agg, NH);
        f32_to_bf16_kernel<<<(NH + 255) / 256, 256, 0, stream>>>(
            (i == 0) ? node_attr : h_f32, h_bf, NH);
        edge_msg_kernel<<<NEDGES / 16 / 8, 256, 0, stream>>>(
            h_bf, pos, edge_attr, src, dst,
            Wt + (size_t)i * HID * IN_PAD, bm + i * HID, agg);
        node_update_kernel<<<(NNODES + 255) / 256, 256, 0, stream>>>(
            agg, Wu + (size_t)i * HID * HID, bu + i * HID,
            ln_g + i * HID, ln_b + i * HID, h_f32, NNODES);
    }

    zero_f32_kernel<<<(NGRAPHS * HID + 255) / 256, 256, 0, stream>>>(gf, NGRAPHS * HID);
    graph_scatter_kernel<<<(NH + 255) / 256, 256, 0, stream>>>(h_f32, graph_ids, gf);
    head_kernel<<<(NGRAPHS + 255) / 256, 256, 0, stream>>>(gf, Wh1, bh1, Wh2, bh2, out);
}
